// Attention_9053791060348
// MI455X (gfx1250) — matrix operand order repaired and checked by reading, not compiled, not run on hardware
//
#include <hip/hip_runtime.h>
#include <cstdint>
#include <cstddef>

#define BB 2
#define SS 2048
#define DDIM 1024
#define HH 16
#define DH 64
#define BS (BB * SS) // 4096

typedef __bf16 bf16x16 __attribute__((ext_vector_type(16)));
typedef __bf16 bf16x4v __attribute__((ext_vector_type(4)));
typedef float  f32x8   __attribute__((ext_vector_type(8)));
typedef float  f32x4v  __attribute__((ext_vector_type(4)));
typedef int    i32x4v  __attribute__((ext_vector_type(4)));

union Frag { uint4 q[2]; bf16x16 v; };

__device__ __forceinline__ f32x8 zero8() {
  f32x8 z = {0.f, 0.f, 0.f, 0.f, 0.f, 0.f, 0.f, 0.f};
  return z;
}

// native f32 -> bf16 (RNE) via hardware cvt
__device__ __forceinline__ unsigned short f2bf(float f) {
  union { __bf16 h; unsigned short u; } v;
  v.h = (__bf16)f;
  return v.u;
}

// packed 4-wide f32 -> bf16 (emits v_cvt_pk_bf16_f32 pairs)
union Cvt4 { bf16x4v v; uint2 u2; unsigned short s[4]; };
__device__ __forceinline__ Cvt4 cvt4(float4 f) {
  f32x4v x = { f.x, f.y, f.z, f.w };
  Cvt4 c;
  c.v = __builtin_convertvector(x, bf16x4v);
  return c;
}

__device__ __forceinline__ f32x8 wmma_bf16(const Frag& a, const Frag& b, f32x8 c) {
  return __builtin_amdgcn_wmma_f32_16x16x32_bf16(false, a.v, false, b.v,
                                                 (short)0, c, false, false);
}

// ---- async global->LDS copy (ASYNCcnt path), builtin if declared else asm ----
__device__ __forceinline__ void async_copy_b128(const void* gsrc, void* ldst) {
#if __has_builtin(__builtin_amdgcn_global_load_async_to_lds_b128)
  __builtin_amdgcn_global_load_async_to_lds_b128(
      (__attribute__((address_space(1))) i32x4v*)gsrc,
      (__attribute__((address_space(3))) i32x4v*)ldst, 0, 0);
#else
  unsigned loff = (unsigned)(unsigned long long)
      (__attribute__((address_space(3))) void*)ldst;
  unsigned long long gaddr = (unsigned long long)gsrc;
  asm volatile("global_load_async_to_lds_b128 %0, %1, off"
               :: "v"(loff), "v"(gaddr) : "memory");
#endif
}

__device__ __forceinline__ void wait_async0() {
#if __has_builtin(__builtin_amdgcn_s_wait_asynccnt)
  __builtin_amdgcn_s_wait_asynccnt(0);
#else
  asm volatile("s_wait_asynccnt 0x0" ::: "memory");
#endif
}

// ---------------------------------------------------------------------------
// Kernel 1: QKV projection.  C[4096 x 64(head)] = X[4096 x 1024] * W[h][1024 x 64]
// Output bf16; vmode 0: [b][h][s][e] (Q scaled, K), vmode 1: [b][h][e][s] (V^T)
// ---------------------------------------------------------------------------
__global__ __launch_bounds__(256) void proj_kernel(
    const float* __restrict__ X, const float* __restrict__ W,
    const float* __restrict__ bias, unsigned short* __restrict__ out,
    float scale, int vmode)
{
  __shared__ unsigned short ldsA[128 * 40]; // [row][k], stride 40 (16B aligned rows)
  __shared__ unsigned short ldsB[64 * 40];  // [n][k] (transposed), stride 40

  const int tid  = threadIdx.x;
  const int w    = tid >> 5;
  const int lane = tid & 31;
  const int hl   = lane >> 4;
  const int l16  = lane & 15;
  const int RM   = blockIdx.x * 128;
  const int h    = blockIdx.y;

  const float* Wb = W + (size_t)h * DDIM * DH;

  f32x8 acc[4];
  #pragma unroll
  for (int j = 0; j < 4; ++j) acc[j] = zero8();

  for (int kb = 0; kb < DDIM; kb += 32) {
    __syncthreads();
    // stage A: 128x32 f32 -> bf16 LDS (packed hardware cvt)
    #pragma unroll
    for (int it = 0; it < 4; ++it) {
      int e = tid + it * 256;          // 1024 float4 slots
      int row = e >> 3, c4 = e & 7;
      float4 f = *(const float4*)(X + (size_t)(RM + row) * DDIM + kb + c4 * 4);
      *(uint2*)&ldsA[row * 40 + c4 * 4] = cvt4(f).u2;
    }
    // stage B: 32(k) x 64(n) f32 -> bf16 LDS transposed [n][k]
    #pragma unroll
    for (int it = 0; it < 2; ++it) {
      int e = tid + it * 256;          // 512 float4 slots
      int rk = e >> 4, c4 = e & 15;
      float4 f = *(const float4*)(Wb + (size_t)(kb + rk) * DH + c4 * 4);
      Cvt4 c = cvt4(f);
      int n = c4 * 4;
      ldsB[(n + 0) * 40 + rk] = c.s[0];
      ldsB[(n + 1) * 40 + rk] = c.s[1];
      ldsB[(n + 2) * 40 + rk] = c.s[2];
      ldsB[(n + 3) * 40 + rk] = c.s[3];
    }
    __syncthreads();
    if (kb + 32 < DDIM)
      __builtin_prefetch(X + (size_t)(RM + (tid >> 1)) * DDIM + kb + 32 + (tid & 1) * 16, 0, 0);

    // A fragment: lane row = w*16 + l16; K run {hl*16..+16}
    Frag a;
    const unsigned short* pa = &ldsA[(w * 16 + l16) * 40 + hl * 16];
    a.q[0] = *(const uint4*)pa;
    a.q[1] = *(const uint4*)(pa + 8);
    // preload all 4 B fragments, then issue the 4 WMMAs back-to-back
    Frag bf[4];
    #pragma unroll
    for (int j = 0; j < 4; ++j) {
      const unsigned short* pb = &ldsB[(j * 16 + l16) * 40 + hl * 16];
      bf[j].q[0] = *(const uint4*)pb;
      bf[j].q[1] = *(const uint4*)(pb + 8);
    }
    #pragma unroll
    for (int j = 0; j < 4; ++j) acc[j] = wmma_bf16(a, bf[j], acc[j]);
  }

  // bias + scale + pack to bf16, scatter to workspace
  #pragma unroll
  for (int j = 0; j < 4; ++j) {
    int nn = j * 16 + l16;
    float bv = bias[(size_t)h * DH + nn];
    #pragma unroll
    for (int r = 0; r < 8; ++r) {
      int m = r + hl * 8;
      int grow = RM + w * 16 + m;
      int bidx = grow >> 11;           // / SS
      int s = grow & (SS - 1);
      float v = (acc[j][r] + bv) * scale;
      size_t idx;
      if (vmode) idx = (((size_t)(bidx * HH + h)) * DH + nn) * SS + s; // V^T
      else       idx = (((size_t)(bidx * HH + h)) * SS + s) * DH + nn; // Q, K
      out[idx] = f2bf(v);
    }
  }
}

// ---------------------------------------------------------------------------
// Kernel 2: causal flash attention.  One 16-query tile per wave, 32-key tiles.
// Q pre-scaled by 1/sqrt(d).  Z out: [b][s][h][e] bf16.
// ---------------------------------------------------------------------------
__global__ __launch_bounds__(128) void attn_kernel(
    const unsigned short* __restrict__ Q, const unsigned short* __restrict__ K,
    const unsigned short* __restrict__ Vt, unsigned short* __restrict__ Z)
{
  __shared__ unsigned short Psh[4 * 16 * 32]; // per-wave 16x32 bf16 P tile

  const int tid  = threadIdx.x;
  const int w    = tid >> 5;
  const int lane = tid & 31;
  const int hl   = lane >> 4;
  const int l16  = lane & 15;
  const int h    = blockIdx.y;
  const int bidx = blockIdx.z;
  const int q0   = blockIdx.x * 64 + w * 16;

  const unsigned short* Qb = Q  + (((size_t)(bidx * HH + h)) * SS) * DH;
  const unsigned short* Kb = K  + (((size_t)(bidx * HH + h)) * SS) * DH;
  const unsigned short* Vb = Vt + (((size_t)(bidx * HH + h)) * DH) * SS;

  // preload Q A-fragments (2 K-chunks over DH=64)
  Frag qa[2];
  const unsigned short* Qrow = Qb + (size_t)(q0 + l16) * DH;
  #pragma unroll
  for (int c = 0; c < 2; ++c) {
    qa[c].q[0] = *(const uint4*)(Qrow + c * 32 + hl * 16);
    qa[c].q[1] = *(const uint4*)(Qrow + c * 32 + 8 + hl * 16);
  }

  float mrow[8], lrow[8];
  f32x8 o[4];
  #pragma unroll
  for (int nt = 0; nt < 4; ++nt) o[nt] = zero8();
  #pragma unroll
  for (int r = 0; r < 8; ++r) { mrow[r] = -1.0e30f; lrow[r] = 0.0f; }

  unsigned short* Pw = &Psh[w * 512];
  const int ntiles = (q0 + 16 + 31) >> 5;

  for (int kt = 0; kt < ntiles; ++kt) {
    const int kbase = kt * 32;
    if (kt + 1 < ntiles) {
      __builtin_prefetch(Kb + (size_t)(kbase + 32 + lane) * DH, 0, 0);
      __builtin_prefetch(Vb + (size_t)(lane * 2) * SS + kbase + 32, 0, 0);
    }

    // scores: S = Q * K^T  (two 16-wide key groups) -- preload all K frags
    Frag kf[2][2];
    #pragma unroll
    for (int c = 0; c < 2; ++c) {
      const unsigned short* Kp0 = Kb + (size_t)(kbase + l16)      * DH + c * 32 + hl * 16;
      const unsigned short* Kp1 = Kb + (size_t)(kbase + 16 + l16) * DH + c * 32 + hl * 16;
      kf[c][0].q[0] = *(const uint4*)Kp0; kf[c][0].q[1] = *(const uint4*)(Kp0 + 8);
      kf[c][1].q[0] = *(const uint4*)Kp1; kf[c][1].q[1] = *(const uint4*)(Kp1 + 8);
    }
    f32x8 s0 = zero8(), s1 = zero8();
    #pragma unroll
    for (int c = 0; c < 2; ++c) {
      s0 = wmma_bf16(qa[c], kf[c][0], s0);
      s1 = wmma_bf16(qa[c], kf[c][1], s1);
    }

    // causal mask + online softmax (rows striped: M = r + 8*hl, N = l16)
    float p0[8], p1[8];
    #pragma unroll
    for (int r = 0; r < 8; ++r) {
      int qg = q0 + r + hl * 8;
      float v0 = (kbase + l16      > qg) ? -1.0e30f : s0[r];
      float v1 = (kbase + 16 + l16 > qg) ? -1.0e30f : s1[r];
      float t = fmaxf(v0, v1);
      t = fmaxf(t, __shfl_xor(t, 1, 32));
      t = fmaxf(t, __shfl_xor(t, 2, 32));
      t = fmaxf(t, __shfl_xor(t, 4, 32));
      t = fmaxf(t, __shfl_xor(t, 8, 32));
      float mn = fmaxf(mrow[r], t);
      float alpha = __expf(mrow[r] - mn);
      mrow[r] = mn;
      p0[r] = __expf(v0 - mn);
      p1[r] = __expf(v1 - mn);
      float rs = p0[r] + p1[r];
      rs += __shfl_xor(rs, 1, 32);
      rs += __shfl_xor(rs, 2, 32);
      rs += __shfl_xor(rs, 4, 32);
      rs += __shfl_xor(rs, 8, 32);
      lrow[r] = lrow[r] * alpha + rs;
      #pragma unroll
      for (int nt = 0; nt < 4; ++nt) o[nt][r] *= alpha;
    }

    // re-stripe P (C-layout -> A-layout) through per-wave LDS tile
    #pragma unroll
    for (int r = 0; r < 8; ++r) {
      int M = r + hl * 8;
      Pw[M * 32 + l16]      = f2bf(p0[r]);
      Pw[M * 32 + 16 + l16] = f2bf(p1[r]);
    }
    asm volatile("s_wait_dscnt 0x0" ::: "memory"); // same-wave LDS RAW fence
    Frag pf;
    const unsigned short* pp = Pw + l16 * 32 + hl * 16;
    pf.q[0] = *(const uint4*)pp;
    pf.q[1] = *(const uint4*)(pp + 8);

    // O += P * V  (V^T layout gives contiguous key runs per lane)
    Frag vf[4];
    #pragma unroll
    for (int nt = 0; nt < 4; ++nt) {
      const unsigned short* Vp = Vb + (size_t)(nt * 16 + l16) * SS + kbase + hl * 16;
      vf[nt].q[0] = *(const uint4*)Vp; vf[nt].q[1] = *(const uint4*)(Vp + 8);
    }
    #pragma unroll
    for (int nt = 0; nt < 4; ++nt) o[nt] = wmma_bf16(pf, vf[nt], o[nt]);
  }

  // normalize and store Z[b][s][h][e]
  #pragma unroll
  for (int r = 0; r < 8; ++r) {
    float inv = 1.0f / lrow[r];
    int s = q0 + r + hl * 8;
    #pragma unroll
    for (int nt = 0; nt < 4; ++nt) {
      int e = nt * 16 + l16;
      size_t idx = (((size_t)bidx * SS + s) * HH + h) * DH + e;
      Z[idx] = f2bf(o[nt][r] * inv);
    }
  }
}

// ---------------------------------------------------------------------------
// Kernel 3: output projection.  out[4096 x 1024] = Z[4096 x 1024] * W_O + b_O
// A-tile staged with ASYNC global->LDS copies (ASYNCcnt path).
// ---------------------------------------------------------------------------
__global__ __launch_bounds__(256) void oproj_kernel(
    const unsigned short* __restrict__ Zb, const float* __restrict__ WO,
    const float* __restrict__ bO, float* __restrict__ Out)
{
  __shared__ unsigned short ldsA[128 * 40];
  __shared__ unsigned short ldsB[64 * 40];

  const int tid  = threadIdx.x;
  const int w    = tid >> 5;
  const int lane = tid & 31;
  const int hl   = lane >> 4;
  const int l16  = lane & 15;
  const int RM   = blockIdx.x * 128;
  const int n0   = blockIdx.y * 64;

  f32x8 acc[4];
  #pragma unroll
  for (int j = 0; j < 4; ++j) acc[j] = zero8();

  const int KD = HH * DH; // 1024
  for (int kb = 0; kb < KD; kb += 32) {
    __syncthreads();
    // stage A: 128x32 bf16 straight copy -> async load to LDS (b128 per slot)
    #pragma unroll
    for (int it = 0; it < 2; ++it) {
      int e = tid + it * 256;          // 512 uint4 slots
      int row = e >> 2, c8 = e & 3;
      async_copy_b128(Zb + (size_t)(RM + row) * KD + kb + c8 * 8,
                      &ldsA[row * 40 + c8 * 8]);
    }
    // stage B: W_O rows k (stride D), transpose into [n][k]
    #pragma unroll
    for (int it = 0; it < 2; ++it) {
      int e = tid + it * 256;
      int rk = e >> 4, c4 = e & 15;
      float4 f = *(const float4*)(WO + (size_t)(kb + rk) * DDIM + n0 + c4 * 4);
      Cvt4 c = cvt4(f);
      int n = c4 * 4;
      ldsB[(n + 0) * 40 + rk] = c.s[0];
      ldsB[(n + 1) * 40 + rk] = c.s[1];
      ldsB[(n + 2) * 40 + rk] = c.s[2];
      ldsB[(n + 3) * 40 + rk] = c.s[3];
    }
    wait_async0();                     // this wave's async LDS writes done
    __syncthreads();                   // ...and visible to all waves
    if (kb + 32 < KD)
      __builtin_prefetch(Zb + (size_t)(RM + (tid >> 1)) * KD + kb + 32 + (tid & 1) * 16, 0, 0);

    Frag a;
    const unsigned short* pa = &ldsA[(w * 16 + l16) * 40 + hl * 16];
    a.q[0] = *(const uint4*)pa;
    a.q[1] = *(const uint4*)(pa + 8);
    Frag bf[4];
    #pragma unroll
    for (int j = 0; j < 4; ++j) {
      const unsigned short* pb = &ldsB[(j * 16 + l16) * 40 + hl * 16];
      bf[j].q[0] = *(const uint4*)pb;
      bf[j].q[1] = *(const uint4*)(pb + 8);
    }
    #pragma unroll
    for (int j = 0; j < 4; ++j) acc[j] = wmma_bf16(a, bf[j], acc[j]);
  }

  #pragma unroll
  for (int j = 0; j < 4; ++j) {
    int nn = n0 + j * 16 + l16;
    float bv = bO[nn];
    #pragma unroll
    for (int r = 0; r < 8; ++r) {
      int grow = RM + w * 16 + r + hl * 8;
      Out[(size_t)grow * DDIM + nn] = acc[j][r] + bv;
    }
  }
}

// ---------------------------------------------------------------------------
extern "C" void kernel_launch(void* const* d_in, const int* in_sizes, int n_in,
                              void* d_out, int out_size, void* d_ws, size_t ws_size,
                              hipStream_t stream) {
  (void)in_sizes; (void)n_in; (void)out_size; (void)ws_size;
  const float* q_in = (const float*)d_in[0];
  const float* k_in = (const float*)d_in[1];
  const float* v_in = (const float*)d_in[2];
  const float* W_Q  = (const float*)d_in[3];
  const float* W_K  = (const float*)d_in[4];
  const float* W_V  = (const float*)d_in[5];
  const float* W_O  = (const float*)d_in[6];
  const float* b_Q  = (const float*)d_in[7];
  const float* b_K  = (const float*)d_in[8];
  const float* b_V  = (const float*)d_in[9];
  const float* b_O  = (const float*)d_in[10];
  float* out = (float*)d_out;

  const size_t HSZ = (size_t)BB * HH * SS * DH; // 4,194,304 bf16 elems = 8 MB
  unsigned short* Qbf = (unsigned short*)d_ws;
  unsigned short* Kbf = Qbf + HSZ;
  unsigned short* Vt  = Kbf + HSZ;
  unsigned short* Zbf = Vt  + HSZ;   // total 32 MB workspace

  dim3 gproj(BS / 128, HH);
  proj_kernel<<<gproj, 256, 0, stream>>>(q_in, W_Q, b_Q, Qbf, 0.125f, 0);
  proj_kernel<<<gproj, 256, 0, stream>>>(k_in, W_K, b_K, Kbf, 1.0f, 0);
  proj_kernel<<<gproj, 256, 0, stream>>>(v_in, W_V, b_V, Vt, 1.0f, 1);

  dim3 gattn(SS / 64, HH, BB);
  attn_kernel<<<gattn, 128, 0, stream>>>(Qbf, Kbf, Vt, Zbf);

  dim3 gop(BS / 128, DDIM / 64);
  oproj_kernel<<<gop, 256, 0, stream>>>(Zbf, W_O, b_O, out);
}
